// AttnBlock_cross_52390011077022
// MI455X (gfx1250) — matrix operand order repaired and checked by reading, not compiled, not run on hardware
//
#include <hip/hip_runtime.h>
#include <hip/hip_bf16.h>
#include <hip/hip_fp16.h>

// ---------------------------------------------------------------------------
// Cross-attention block (B=4, C=256, H=W=64, G=32) for gfx1250 (MI455X).
// Flash-attention on v_wmma_f32_16x16x32_f16 with double-buffered async
// global->LDS staging (ASYNCcnt) for K and channel-major V.
// ---------------------------------------------------------------------------

typedef __attribute__((ext_vector_type(16))) _Float16 v16h;
typedef __attribute__((ext_vector_type(8)))  _Float16 v8h;
typedef __attribute__((ext_vector_type(8)))  float    v8f;
typedef __attribute__((ext_vector_type(4)))  float    v4f;

#define NPOS 4096   // H*W
#define CCH  256    // channels
#define NB   4      // batch
#define KT   32     // key tile for flash attention

// ---- WMMA wrapper ---------------------------------------------------------
__device__ __forceinline__ v8f wmma_f16(v16h a, v16h b, v8f c) {
  return __builtin_amdgcn_wmma_f32_16x16x32_f16(
      /*neg_a=*/false, a, /*neg_b=*/false, b,
      /*c_mod=*/(short)0, c, /*reuse_a=*/false, /*reuse_b=*/false);
}

// ---- A fragment: 16x32 f16, row-major source with row stride rs (elems) ---
// lane L -> row L%16 ; elems 0..7 = K (L/16)*16 .. +7 ;
// elems 8..15 = K (L/16)*16+8 .. +15  (two 16B chunks)
__device__ __forceinline__ v16h load_a_frag(const _Float16* base, int rs, int lane) {
  const _Float16* p = base + (lane & 15) * rs + ((lane >> 4) << 4);
  v8h lo = *(const v8h*)(p);
  v8h hi = *(const v8h*)(p + 8);
  v16h a;
#pragma unroll
  for (int i = 0; i < 8; ++i) { a[i] = lo[i]; a[8 + i] = hi[i]; }
  return a;
}

// ---- B fragment: 32x16 f16 from K-contiguous (transposed) storage ---------
// lane L -> col L%16 ; K = (L/16)*16 .. +15  (one 32B chunk per lane)
__device__ __forceinline__ v16h load_b_frag(const _Float16* base, int rs, int lane) {
  return *(const v16h*)(base + (lane & 15) * rs + ((lane >> 4) << 4));
}

// ---------------------------------------------------------------------------
// Kernel 0: transpose weights [c][d] f32 -> [d][c] f16
// ---------------------------------------------------------------------------
__global__ void prep_weights(const float* __restrict__ W0, const float* __restrict__ W1,
                             const float* __restrict__ W2, const float* __restrict__ W3,
                             _Float16* __restrict__ wt) {
  int m = blockIdx.y;
  const float* src = (m == 0) ? W0 : (m == 1) ? W1 : (m == 2) ? W2 : W3;
  int i = blockIdx.x * 256 + threadIdx.x;       // 0..65535
  int c = i >> 8, d = i & 255;
  wt[(size_t)m * 65536 + d * 256 + c] = (_Float16)src[c * 256 + d];
}

// ---------------------------------------------------------------------------
// Kernel 1: GroupNorm  (x -> hb, cond -> fb), output row-major [n][c] f16.
// Phase 2 writes one packed b128 per position (group = 8 contiguous chans).
// ---------------------------------------------------------------------------
__global__ __launch_bounds__(256)
void gn_kernel(const float* __restrict__ x, const float* __restrict__ cond,
               const float* __restrict__ gamma, const float* __restrict__ beta,
               _Float16* __restrict__ hb, _Float16* __restrict__ fb) {
  int bg = blockIdx.x;                 // B*G = 128
  int which = blockIdx.y;              // 0: x, 1: cond
  int b = bg >> 5, g = bg & 31;
  const float* src = (which ? cond : x) + (size_t)(b * CCH + g * 8) * NPOS;
  _Float16* dst = (which ? fb : hb) + (size_t)b * NPOS * CCH;

  __shared__ float red[20];
  float s = 0.f, ss = 0.f;
  for (int i = threadIdx.x; i < 8 * NPOS; i += 256) {
    float v = src[i]; s += v; ss += v * v;
  }
#pragma unroll
  for (int m = 16; m >= 1; m >>= 1) { s += __shfl_xor(s, m, 32); ss += __shfl_xor(ss, m, 32); }
  int wid = threadIdx.x >> 5;
  if ((threadIdx.x & 31) == 0) { red[wid] = s; red[8 + wid] = ss; }
  __syncthreads();
  if (threadIdx.x == 0) {
    float a = 0.f, bb = 0.f;
    for (int i = 0; i < 8; ++i) { a += red[i]; bb += red[8 + i]; }
    red[16] = a; red[17] = bb;
  }
  __syncthreads();
  const float inv_n = 1.0f / (8.0f * NPOS);
  float mu   = red[16] * inv_n;
  float var  = red[17] * inv_n - mu * mu;
  float rinv = rsqrtf(var + 1e-6f);

  float gm[8], bt[8];
#pragma unroll
  for (int cl = 0; cl < 8; ++cl) { gm[cl] = gamma[g * 8 + cl] * rinv; bt[cl] = beta[g * 8 + cl]; }

  for (int n = threadIdx.x; n < NPOS; n += 256) {
    v8h pk;
#pragma unroll
    for (int cl = 0; cl < 8; ++cl)
      pk[cl] = (_Float16)((src[(size_t)cl * NPOS + n] - mu) * gm[cl] + bt[cl]);
    *(v8h*)(dst + (size_t)n * CCH + g * 8) = pk;
  }
}

// ---------------------------------------------------------------------------
// Kernel 2: Q/K/V projections.  z=0: Q=(hb*W0+b0)/16 row-major [n][c]
//           z=1: K row-major [n][c]
//           z=2: V channel-major [c][n]  (single packed b128 store per tile)
// ---------------------------------------------------------------------------
__global__ __launch_bounds__(128)
void qkv_kernel(const _Float16* __restrict__ hb, const _Float16* __restrict__ fb,
                const _Float16* __restrict__ wt,
                const float* __restrict__ b0, const float* __restrict__ b1,
                const float* __restrict__ b2,
                _Float16* __restrict__ Qb, _Float16* __restrict__ Kb,
                _Float16* __restrict__ Vt) {
  int z = blockIdx.z;
  const _Float16* X  = (z == 0 ? hb : fb) + (size_t)blockIdx.y * NPOS * CCH;
  const _Float16* Wm = wt + (size_t)z * 65536;
  const float* bias  = (z == 0) ? b0 : (z == 1) ? b1 : b2;
  _Float16* Y = ((z == 0) ? Qb : (z == 1) ? Kb : Vt) + (size_t)blockIdx.y * NPOS * CCH;
  const float sc = (z == 0) ? 0.0625f : 1.0f;   // fold Ch^-0.5 into Q

  int lane = threadIdx.x & 31;
  int wv   = threadIdx.x >> 5;
  int row0 = blockIdx.x * 64 + wv * 16;

  v8f zero = {0.f, 0.f, 0.f, 0.f, 0.f, 0.f, 0.f, 0.f};

  v16h af[8];
#pragma unroll
  for (int ks = 0; ks < 8; ++ks)
    af[ks] = load_a_frag(X + (size_t)row0 * CCH + ks * 32, CCH, lane);

#pragma unroll
  for (int t = 0; t < 16; ++t) {
    v8f acc = zero;
#pragma unroll
    for (int ks = 0; ks < 8; ++ks) {
      v16h bf = load_b_frag(Wm + (size_t)t * 16 * CCH + ks * 32, CCH, lane);
      acc = wmma_f16(af[ks], bf, acc);
    }
    int col = t * 16 + (lane & 15);
    float bc = bias[col];
    if (z == 2) {
      // channel-major: entries r are 8 consecutive positions -> one b128
      v8h pk;
#pragma unroll
      for (int r = 0; r < 8; ++r) pk[r] = (_Float16)(acc[r] + bc);
      *(v8h*)(Y + (size_t)col * NPOS + row0 + 8 * (lane >> 4)) = pk;
    } else {
#pragma unroll
      for (int r = 0; r < 8; ++r) {
        int row = row0 + r + 8 * (lane >> 4);
        Y[(size_t)row * CCH + col] = (_Float16)((acc[r] + bc) * sc);
      }
    }
  }
}

// ---------------------------------------------------------------------------
// Kernel 3: flash attention.  block = 256 thr (8 waves).
// Wave pair (qg) shares 16 queries; each wave owns 128 channels of O.
// K (row-major) and V (channel-major) tiles staged via DOUBLE-BUFFERED
// async global->LDS b128 copies; prefetch of tile j+1 overlaps compute on j.
// ---------------------------------------------------------------------------
__global__ __launch_bounds__(256)
void attn_kernel(const _Float16* __restrict__ Qb, const _Float16* __restrict__ Kb,
                 const _Float16* __restrict__ Vt, _Float16* __restrict__ Ab) {
  int b = blockIdx.y;
  int qbase = blockIdx.x * 64;
  int lane = threadIdx.x & 31;
  int wv   = threadIdx.x >> 5;  // 0..7
  int qg   = wv >> 1;           // query group 0..3 (16 queries each)
  int half = wv & 1;            // channel half: cols half*128 .. +127
  const _Float16* Q = Qb + ((size_t)b * NPOS + qbase + qg * 16) * CCH;
  const _Float16* K = Kb + (size_t)b * NPOS * CCH;
  const _Float16* V = Vt + (size_t)b * NPOS * CCH;   // channel-major [c][n]

  __shared__ __align__(32) _Float16 lK[2][KT][CCH];    // 32 KB, row-major keys
  __shared__ __align__(32) _Float16 lV[2][CCH][KT];    // 32 KB, chan-major V
  __shared__ __align__(32) _Float16 lPh[8][16][KT];    //  8 KB, per-wave P f16

  // preload Q A-fragments (16 queries x 256 ch -> 8 fragments, 64 VGPRs)
  v16h qf[8];
#pragma unroll
  for (int ks = 0; ks < 8; ++ks) qf[ks] = load_a_frag(Q + ks * 32, CCH, lane);

  v8f zero = {0.f, 0.f, 0.f, 0.f, 0.f, 0.f, 0.f, 0.f};
  v8f o[8];
#pragma unroll
  for (int t = 0; t < 8; ++t) o[t] = zero;
  float m_[8], l_[8];
#pragma unroll
  for (int r = 0; r < 8; ++r) { m_[r] = -1e30f; l_[r] = 0.f; }

  // ---- async stage of one 32-key tile into buffer nb ----
  auto stage = [&](int j, int nb) {
    uint32_t ldsKb = (uint32_t)(uintptr_t)&lK[nb][0][0];
    uint32_t ldsVb = (uint32_t)(uintptr_t)&lV[nb][0][0];
    for (int i = threadIdx.x; i < KT * CCH / 8; i += 256) {
      int row = i >> 5, c8 = (i & 31) * 8;       // K: 32 rows x 32 chunks
      asm volatile("global_load_async_to_lds_b128 %0, %1, off"
                   :: "v"(ldsKb + (uint32_t)((row * CCH + c8) * 2)),
                      "v"((uint64_t)(uintptr_t)(K + (size_t)(j + row) * CCH + c8))
                   : "memory");
      int vc = i >> 2, k8 = (i & 3) * 8;         // V: 256 rows x 4 chunks
      asm volatile("global_load_async_to_lds_b128 %0, %1, off"
                   :: "v"(ldsVb + (uint32_t)((vc * KT + k8) * 2)),
                      "v"((uint64_t)(uintptr_t)(V + (size_t)vc * NPOS + j + k8))
                   : "memory");
    }
  };

  stage(0, 0);   // prologue

  for (int j0 = 0; j0 < NPOS; j0 += KT) {
    int buf = (j0 >> 5) & 1;
    asm volatile("s_wait_asynccnt 0x0" ::: "memory");  // my stage(buf) done
    __syncthreads();                                   // everyone's done
    if (j0 + KT < NPOS) stage(j0 + KT, buf ^ 1);       // prefetch next tile

    // S = Q K^T : 16 queries x 32 keys, K-dim 256 in 8 steps
    v8f s0 = zero, s1 = zero;
#pragma unroll
    for (int ks = 0; ks < 8; ++ks) {
      v16h kb0 = load_b_frag(&lK[buf][0][ks * 32],  CCH, lane);
      v16h kb1 = load_b_frag(&lK[buf][16][ks * 32], CCH, lane);
      s0 = wmma_f16(qf[ks], kb0, s0);
      s1 = wmma_f16(qf[ks], kb1, s1);
    }

    // online softmax (row = r + 8*(lane/16), col = lane%16); P stored f16
    float alpha[8];
#pragma unroll
    for (int r = 0; r < 8; ++r) {
      float t = fmaxf(s0[r], s1[r]);
#pragma unroll
      for (int msk = 8; msk >= 1; msk >>= 1) t = fmaxf(t, __shfl_xor(t, msk, 32));
      float mn = fmaxf(m_[r], t);
      alpha[r] = __expf(m_[r] - mn);
      m_[r] = mn;
      float p0 = __expf(s0[r] - mn);
      float p1 = __expf(s1[r] - mn);
      int rr = r + 8 * (lane >> 4);
      lPh[wv][rr][lane & 15]        = (_Float16)p0;
      lPh[wv][rr][16 + (lane & 15)] = (_Float16)p1;
      float ps = p0 + p1;
#pragma unroll
      for (int msk = 8; msk >= 1; msk >>= 1) ps += __shfl_xor(ps, msk, 32);
      l_[r] = l_[r] * alpha[r] + ps;
    }
    asm volatile("s_wait_dscnt 0x0" ::: "memory");  // wave-local LDS visibility

    // P is already in A-fragment-friendly row-major f16 layout
    v16h pa = load_a_frag(&lPh[wv][0][0], KT, lane);

    // O = O*alpha + P V  for this wave's 8 channel tiles
#pragma unroll
    for (int t = 0; t < 8; ++t) {
#pragma unroll
      for (int r = 0; r < 8; ++r) o[t][r] *= alpha[r];
      v16h vb = load_b_frag(&lV[buf][half * 128 + t * 16][0], KT, lane);
      o[t] = wmma_f16(pa, vb, o[t]);
    }
  }

  // epilogue: normalize and store row-major [n][c] f16
#pragma unroll
  for (int t = 0; t < 8; ++t) {
    int col = half * 128 + t * 16 + (lane & 15);
#pragma unroll
    for (int r = 0; r < 8; ++r) {
      int row = qbase + qg * 16 + r + 8 * (lane >> 4);
      Ab[((size_t)b * NPOS + row) * CCH + col] = (_Float16)(o[t][r] / l_[r]);
    }
  }
}

// ---------------------------------------------------------------------------
// Kernel 4: out = x + A*W3 + b3, stored in reference layout [b][c][n] f32.
// Per-lane entries are 8 consecutive positions -> packed b128 load/store.
// ---------------------------------------------------------------------------
__global__ __launch_bounds__(128)
void out_kernel(const _Float16* __restrict__ Ab, const _Float16* __restrict__ wt,
                const float* __restrict__ b3, const float* __restrict__ x,
                float* __restrict__ out) {
  int b = blockIdx.y;
  int lane = threadIdx.x & 31, wv = threadIdx.x >> 5;
  int row0 = blockIdx.x * 64 + wv * 16;
  const _Float16* A   = Ab + ((size_t)b * NPOS + row0) * CCH;
  const _Float16* W3t = wt + (size_t)3 * 65536;

  v8f zero = {0.f, 0.f, 0.f, 0.f, 0.f, 0.f, 0.f, 0.f};

  v16h af[8];
#pragma unroll
  for (int ks = 0; ks < 8; ++ks)
    af[ks] = load_a_frag(A + ks * 32, CCH, lane);

#pragma unroll
  for (int t = 0; t < 16; ++t) {
    v8f acc = zero;
#pragma unroll
    for (int ks = 0; ks < 8; ++ks) {
      v16h bf = load_b_frag(W3t + (size_t)t * 16 * CCH + ks * 32, CCH, lane);
      acc = wmma_f16(af[ks], bf, acc);
    }
    int col = t * 16 + (lane & 15);
    float bc = b3[col];
    size_t idx0 = ((size_t)b * CCH + col) * NPOS + row0 + 8 * (lane >> 4);
    v4f x0 = *(const v4f*)(x + idx0);
    v4f x1 = *(const v4f*)(x + idx0 + 4);
    v4f o0, o1;
#pragma unroll
    for (int r = 0; r < 4; ++r) {
      o0[r] = x0[r] + acc[r]     + bc;
      o1[r] = x1[r] + acc[4 + r] + bc;
    }
    *(v4f*)(out + idx0)     = o0;
    *(v4f*)(out + idx0 + 4) = o1;
  }
}

// ---------------------------------------------------------------------------
extern "C" void kernel_launch(void* const* d_in, const int* in_sizes, int n_in,
                              void* d_out, int out_size, void* d_ws, size_t ws_size,
                              hipStream_t stream) {
  const float* x     = (const float*)d_in[0];
  const float* cond  = (const float*)d_in[1];
  const float* gamma = (const float*)d_in[2];
  const float* beta  = (const float*)d_in[3];
  const float* W0 = (const float*)d_in[4];
  const float* b0 = (const float*)d_in[5];
  const float* W1 = (const float*)d_in[6];
  const float* b1 = (const float*)d_in[7];
  const float* W2 = (const float*)d_in[8];
  const float* b2 = (const float*)d_in[9];
  const float* W3 = (const float*)d_in[10];
  const float* b3 = (const float*)d_in[11];
  float* out = (float*)d_out;

  // workspace carve-up (f16 elements): 4 transposed weights + 6 activation bufs
  const size_t BNC = (size_t)NB * NPOS * CCH;   // 4,194,304 halves each
  _Float16* ws = (_Float16*)d_ws;
  _Float16* wt = ws;                  // 4 * 65536
  _Float16* hb = wt + (size_t)4 * 65536;
  _Float16* fb = hb + BNC;
  _Float16* Qb = fb + BNC;
  _Float16* Kb = Qb + BNC;
  _Float16* Vt = Kb + BNC;            // channel-major [b][c][n]
  _Float16* Ab = Vt + BNC;            // total ~51 MB

  prep_weights<<<dim3(256, 4), 256, 0, stream>>>(W0, W1, W2, W3, wt);
  gn_kernel<<<dim3(128, 2), 256, 0, stream>>>(x, cond, gamma, beta, hb, fb);
  qkv_kernel<<<dim3(64, NB, 3), 128, 0, stream>>>(hb, fb, wt, b0, b1, b2, Qb, Kb, Vt);
  attn_kernel<<<dim3(NPOS / 64, NB), 256, 0, stream>>>(Qb, Kb, Vt, Ab);
  out_kernel<<<dim3(NPOS / 64, NB), 128, 0, stream>>>(Ab, wt, b3, x, out);
}
